// SelfAttention_39943195853527
// MI455X (gfx1250) — matrix operand order repaired and checked by reading, not compiled, not run on hardware
//
#include <hip/hip_runtime.h>

// ---------------- types ----------------
typedef __attribute__((ext_vector_type(16))) __bf16 v16bf;
typedef __attribute__((ext_vector_type(8)))  float  v8f;
typedef __attribute__((ext_vector_type(8)))  unsigned int v8u;

static __device__ __forceinline__ unsigned short f2bf(float f) {
    unsigned int u = __builtin_bit_cast(unsigned int, f);
    u += 0x7FFFu + ((u >> 16) & 1u);           // round-to-nearest-even
    return (unsigned short)(u >> 16);
}

static __device__ __forceinline__ v16bf make_v16bf(uint4 a, uint4 b) {
    v8u t;
    t[0] = a.x; t[1] = a.y; t[2] = a.z; t[3] = a.w;
    t[4] = b.x; t[5] = b.y; t[6] = b.z; t[7] = b.w;
    return __builtin_bit_cast(v16bf, t);
}

// A-fragment (16x32 bf16): lane half selects K groups {0-7,16-23} / {8-15,24-31}
static __device__ __forceinline__ v16bf frag_a(const unsigned short* lds, int row, int kk, int half) {
    const unsigned short* base = lds + row * 72 + kk + 8 * half;
    uint4 lo = *(const uint4*)(base);
    uint4 hi = *(const uint4*)(base + 16);
    return make_v16bf(lo, hi);
}

// B-fragment (32x16 bf16): lane half selects K groups {0-7,16-23} / {8-15,24-31} of column (lane&15)
static __device__ __forceinline__ v16bf frag_b(const unsigned short* lds, int row, int kk, int half) {
    const unsigned short* base = lds + row * 72 + kk + 8 * half;
    uint4 lo = *(const uint4*)(base);
    uint4 hi = *(const uint4*)(base + 16);
    return make_v16bf(lo, hi);
}

static __device__ __forceinline__ v8f wmma_bf16(v16bf a, v16bf b, v8f c) {
    return __builtin_amdgcn_wmma_f32_16x16x32_bf16(false, a, false, b, (short)0, c, false, false);
}

// async global -> LDS (128-bit), tracked by ASYNCcnt
static __device__ __forceinline__ void async_b128(const void* lds_ptr, const void* gptr) {
    unsigned lds = (unsigned)(uintptr_t)lds_ptr;           // low 32 bits of shared flat addr = LDS offset
    unsigned long long g = (unsigned long long)(uintptr_t)gptr;
    asm volatile("global_load_async_to_lds_b128 %0, %1, off" :: "v"(lds), "v"(g) : "memory");
}
static __device__ __forceinline__ void wait_async0() {
    asm volatile("s_wait_asynccnt 0x0" ::: "memory");
}

// ---------------- fp32 -> bf16 bulk convert ----------------
__global__ __launch_bounds__(256) void f32_to_bf16(const float4* __restrict__ in,
                                                   uint2* __restrict__ out, int n4) {
    int i = blockIdx.x * 256 + threadIdx.x;
    if (i < n4) {
        float4 v = in[i];
        uint2 st;
        st.x = (unsigned)f2bf(v.x) | ((unsigned)f2bf(v.y) << 16);
        st.y = (unsigned)f2bf(v.z) | ((unsigned)f2bf(v.w) << 16);
        out[i] = st;
    }
}

// ---------------- GEMM: C[M,N] = A[M,K] . W[N,K]^T (+bias)(+resid)(relu) ----------------
// bf16 inputs, fp32 accum. 128x128 tile, BK=64, 256 threads = 8 waves (wave32).
// Double-buffered LDS fed by async global->LDS b128 loads.
__global__ __launch_bounds__(256) void gemm_bf16_wmma(
    const unsigned short* __restrict__ A, const unsigned short* __restrict__ W,
    const float* __restrict__ bias, const float* __restrict__ resid,
    float* __restrict__ outf, unsigned short* __restrict__ outbf,
    int M, int N, int K, int relu)
{
    __shared__ unsigned short As[2][128 * 72];
    __shared__ unsigned short Bs[2][128 * 72];

    const int t    = threadIdx.x;
    const int m0   = blockIdx.y * 128;
    const int n0   = blockIdx.x * 128;
    const int wid  = t >> 5;
    const int lane = t & 31;
    const int half = lane >> 4;
    const int cl   = lane & 15;
    const int wm   = wid >> 1;   // 0..3
    const int wn   = wid & 1;    // 0..1

    v8f acc[2][4];
#pragma unroll
    for (int mi = 0; mi < 2; ++mi)
#pragma unroll
        for (int ni = 0; ni < 4; ++ni)
            acc[mi][ni] = (v8f){0.f, 0.f, 0.f, 0.f, 0.f, 0.f, 0.f, 0.f};

    auto issue_tiles = [&](int buf, int k0) {
#pragma unroll
        for (int i = 0; i < 4; ++i) {
            int idx = t + i * 256;              // 1024 x b128 per tile
            int row = idx >> 3, c8 = idx & 7;
            async_b128(&As[buf][row * 72 + c8 * 8], A + (size_t)(m0 + row) * K + k0 + c8 * 8);
            async_b128(&Bs[buf][row * 72 + c8 * 8], W + (size_t)(n0 + row) * K + k0 + c8 * 8);
        }
    };

    issue_tiles(0, 0);
    wait_async0();
    __syncthreads();

    const int KT = K >> 6;
    for (int kt = 0; kt < KT; ++kt) {
        const int cur = kt & 1;
        if (kt + 1 < KT) issue_tiles(cur ^ 1, (kt + 1) * 64);

        const unsigned short* Ac = As[cur];
        const unsigned short* Bc = Bs[cur];
#pragma unroll
        for (int ks = 0; ks < 2; ++ks) {
            int kk = ks * 32;
            v16bf a0 = frag_a(Ac, wm * 32 + cl,      kk, half);
            v16bf a1 = frag_a(Ac, wm * 32 + 16 + cl, kk, half);
#pragma unroll
            for (int ni = 0; ni < 4; ++ni) {
                v16bf b = frag_b(Bc, wn * 64 + ni * 16 + cl, kk, half);
                acc[0][ni] = wmma_bf16(a0, b, acc[0][ni]);
                acc[1][ni] = wmma_bf16(a1, b, acc[1][ni]);
            }
        }
        wait_async0();       // next buffer's async loads done (this wave)
        __syncthreads();     // all waves done computing cur + loading next
    }

    // ---- epilogue ----
#pragma unroll
    for (int mi = 0; mi < 2; ++mi) {
#pragma unroll
        for (int ni = 0; ni < 4; ++ni) {
            int gcol = n0 + wn * 64 + ni * 16 + cl;
            float bi = bias ? bias[gcol] : 0.0f;
#pragma unroll
            for (int r = 0; r < 8; ++r) {
                int grow = m0 + wm * 32 + mi * 16 + r + 8 * half;
                float v = acc[mi][ni][r] + bi;
                if (resid) v += resid[grow * N + gcol];
                if (relu)  v  = fmaxf(v, 0.0f);
                if (outf)  outf[grow * N + gcol]  = v;
                if (outbf) outbf[grow * N + gcol] = f2bf(v);
            }
        }
    }
}

// ---------------- flash attention (q=k=v = p), bf16 WMMA, online softmax ----------------
// grid: (S/64, B*NH), 128 threads = 4 waves; wave w owns 16 q rows. Output bf16.
__global__ __launch_bounds__(128) void attn_kernel(
    const unsigned short* __restrict__ p,    // bf16 [B,S,HID]
    const unsigned char*  __restrict__ mask, // [B,S] true = masked out
    unsigned short* __restrict__ att)        // bf16 [B,S,HID]
{
    constexpr int S = 2048, HID = 1024, NH = 16;
    __shared__ unsigned short Qs[64 * 72];
    __shared__ unsigned short Ks[64 * 72];
    __shared__ unsigned short Vt[64 * 72];   // transposed: Vt[d][key]
    __shared__ unsigned short Ps[64 * 72];
    __shared__ float maskadd[64];

    const int t    = threadIdx.x;
    const int b    = blockIdx.y >> 4;
    const int h    = blockIdx.y & (NH - 1);
    const int q0   = blockIdx.x * 64;
    const int w    = t >> 5;
    const int lane = t & 31;
    const int half = lane >> 4;
    const int cl   = lane & 15;
    const unsigned short* ph = p + (size_t)b * S * HID + h * 64;

    // async-load Q tile once (64 x 64 bf16); completion covered by first loop wait
#pragma unroll
    for (int i = 0; i < 4; ++i) {
        int idx = t + i * 128;               // 512 x b128
        int row = idx >> 3, c8 = idx & 7;
        async_b128(&Qs[row * 72 + c8 * 8], ph + (q0 + row) * HID + c8 * 8);
    }

    float mrow[8], lrow[8];
#pragma unroll
    for (int r = 0; r < 8; ++r) { mrow[r] = -3.0e38f; lrow[r] = 0.0f; }
    v8f o[4];
#pragma unroll
    for (int ni = 0; ni < 4; ++ni) o[ni] = (v8f){0.f,0.f,0.f,0.f,0.f,0.f,0.f,0.f};

    for (int kv0 = 0; kv0 < S; kv0 += 64) {
        // async-load K tile (also V data); mask additive terms
#pragma unroll
        for (int i = 0; i < 4; ++i) {
            int idx = t + i * 128;
            int row = idx >> 3, c8 = idx & 7;
            async_b128(&Ks[row * 72 + c8 * 8], ph + (kv0 + row) * HID + c8 * 8);
        }
        if (t < 64) maskadd[t] = mask[b * S + kv0 + t] ? -1.0e9f : 0.0f;
        wait_async0();
        __syncthreads();
        // transpose into Vt[d][key]
#pragma unroll
        for (int i = 0; i < 32; ++i) {
            int idx = t * 32 + i;
            int key = idx >> 6, d = idx & 63;
            Vt[d * 72 + key] = Ks[key * 72 + d];
        }
        __syncthreads();

        // S = Q K^T  (contract over d = 64 -> 2 wmma k-steps)
        v8f s[4];
#pragma unroll
        for (int ni = 0; ni < 4; ++ni) s[ni] = (v8f){0.f,0.f,0.f,0.f,0.f,0.f,0.f,0.f};
#pragma unroll
        for (int ks = 0; ks < 2; ++ks) {
            int kk = ks * 32;
            v16bf a = frag_a(Qs, w * 16 + cl, kk, half);
#pragma unroll
            for (int ni = 0; ni < 4; ++ni) {
                v16bf bb = frag_b(Ks, ni * 16 + cl, kk, half);
                s[ni] = wmma_bf16(a, bb, s[ni]);
            }
        }

        float madd_r[4];
#pragma unroll
        for (int ni = 0; ni < 4; ++ni) madd_r[ni] = maskadd[ni * 16 + cl];

        // online softmax per row (rows live in 16-lane halves)
#pragma unroll
        for (int r = 0; r < 8; ++r) {
            float rm = -3.0e38f;
#pragma unroll
            for (int ni = 0; ni < 4; ++ni) {
                float val = s[ni][r] * 0.125f + madd_r[ni];
                s[ni][r] = val;
                rm = fmaxf(rm, val);
            }
#pragma unroll
            for (int off = 8; off > 0; off >>= 1)
                rm = fmaxf(rm, __shfl_xor(rm, off, 32));
            float mnew  = fmaxf(mrow[r], rm);
            float alpha = __expf(mrow[r] - mnew);
            mrow[r] = mnew;
            float rs = 0.0f;
#pragma unroll
            for (int ni = 0; ni < 4; ++ni) {
                float pv = __expf(s[ni][r] - mnew);
                s[ni][r] = pv;
                rs += pv;
            }
#pragma unroll
            for (int off = 8; off > 0; off >>= 1)
                rs += __shfl_xor(rs, off, 32);
            lrow[r] = lrow[r] * alpha + rs;
#pragma unroll
            for (int ni = 0; ni < 4; ++ni) o[ni][r] *= alpha;
            int ql = w * 16 + r + 8 * half;   // rows private to this wave -> DS in-order, no barrier
#pragma unroll
            for (int ni = 0; ni < 4; ++ni)
                Ps[ql * 72 + ni * 16 + cl] = f2bf(s[ni][r]);
        }

        // O += P V  (contract over key = 64)
#pragma unroll
        for (int ks = 0; ks < 2; ++ks) {
            int kk = ks * 32;
            v16bf a = frag_a(Ps, w * 16 + cl, kk, half);
#pragma unroll
            for (int ni = 0; ni < 4; ++ni) {
                v16bf bb = frag_b(Vt, ni * 16 + cl, kk, half);
                o[ni] = wmma_bf16(a, bb, o[ni]);
            }
        }
        __syncthreads();
    }

    // normalize and write out bf16 [B,S,HID]
#pragma unroll
    for (int ni = 0; ni < 4; ++ni) {
#pragma unroll
        for (int r = 0; r < 8; ++r) {
            int q = q0 + w * 16 + r + 8 * half;
            int d = ni * 16 + cl;
            att[((size_t)b * S + q) * HID + h * 64 + d] = f2bf(o[ni][r] / lrow[r]);
        }
    }
}

// ---------------- LayerNorm over last dim (1024), block per row; optional bf16 copy ----------------
__global__ __launch_bounds__(256) void ln_kernel(
    const float* __restrict__ x, const float* __restrict__ g,
    const float* __restrict__ be, float* __restrict__ out,
    unsigned short* __restrict__ outb)
{
    __shared__ float red[8];
    const int t = threadIdx.x;
    const int row = blockIdx.x;
    const float4* x4 = (const float4*)(x + (size_t)row * 1024);
    float4 v = x4[t];

    float s = v.x + v.y + v.z + v.w;
#pragma unroll
    for (int off = 16; off > 0; off >>= 1) s += __shfl_xor(s, off, 32);
    if ((t & 31) == 0) red[t >> 5] = s;
    __syncthreads();
    if (t == 0) { float a = 0; for (int i = 0; i < 8; ++i) a += red[i]; red[0] = a; }
    __syncthreads();
    float mean = red[0] * (1.0f / 1024.0f);
    __syncthreads();

    float dx = v.x - mean, dy = v.y - mean, dz = v.z - mean, dw = v.w - mean;
    float s2 = dx * dx + dy * dy + dz * dz + dw * dw;
#pragma unroll
    for (int off = 16; off > 0; off >>= 1) s2 += __shfl_xor(s2, off, 32);
    if ((t & 31) == 0) red[t >> 5] = s2;
    __syncthreads();
    if (t == 0) { float a = 0; for (int i = 0; i < 8; ++i) a += red[i]; red[0] = a; }
    __syncthreads();
    float inv = rsqrtf(red[0] * (1.0f / 1024.0f) + 1e-6f);

    const float4 gv = ((const float4*)g)[t];
    const float4 bv = ((const float4*)be)[t];
    float4 ov;
    ov.x = gv.x * dx * inv + bv.x;
    ov.y = gv.y * dy * inv + bv.y;
    ov.z = gv.z * dz * inv + bv.z;
    ov.w = gv.w * dw * inv + bv.w;
    ((float4*)(out + (size_t)row * 1024))[t] = ov;
    if (outb) {
        uint2 st;
        st.x = (unsigned)f2bf(ov.x) | ((unsigned)f2bf(ov.y) << 16);
        st.y = (unsigned)f2bf(ov.z) | ((unsigned)f2bf(ov.w) << 16);
        ((uint2*)(outb + (size_t)row * 1024))[t] = st;
    }
}

// ---------------- launch ----------------
extern "C" void kernel_launch(void* const* d_in, const int* in_sizes, int n_in,
                              void* d_out, int out_size, void* d_ws, size_t ws_size,
                              hipStream_t stream) {
    constexpr int B = 4, S = 2048, HID = 1024, NH = 16, FF = 4096;
    constexpr int M = B * S;  // 8192

    const float*         x    = (const float*)d_in[0];
    const unsigned char* mask = (const unsigned char*)d_in[1];
    const float* Wv = (const float*)d_in[2];
    const float* bv = (const float*)d_in[3];
    const float* Wm = (const float*)d_in[4];
    const float* bm = (const float*)d_in[5];
    const float* W1 = (const float*)d_in[6];
    const float* b1 = (const float*)d_in[7];
    const float* W2 = (const float*)d_in[8];
    const float* b2 = (const float*)d_in[9];
    const float* g1  = (const float*)d_in[10];
    const float* be1 = (const float*)d_in[11];
    const float* g2  = (const float*)d_in[12];
    const float* be2 = (const float*)d_in[13];

    char* ws = (char*)d_ws;
    size_t off = 0;
    auto take = [&](size_t bytes) { void* p = ws + off; off += bytes; return p; };
    unsigned short* x_bf  = (unsigned short*)take((size_t)M * HID * 2);   // also reused for att_bf
    unsigned short* p_bf  = (unsigned short*)take((size_t)M * HID * 2);
    float*          h1    = (float*)take((size_t)M * HID * 4);            // also reused for h2
    float*          x1    = (float*)take((size_t)M * HID * 4);
    unsigned short* x1_bf = (unsigned short*)take((size_t)M * HID * 2);
    unsigned short* ff1   = (unsigned short*)take((size_t)M * FF * 2);
    unsigned short* Wv_bf = (unsigned short*)take((size_t)HID * HID * 2);
    unsigned short* Wm_bf = (unsigned short*)take((size_t)HID * HID * 2);
    unsigned short* W1_bf = (unsigned short*)take((size_t)FF * HID * 2);
    unsigned short* W2_bf = (unsigned short*)take((size_t)FF * HID * 2);
    unsigned short* att_bf = x_bf;   // x_bf dead after GEMM1
    float*          h2     = h1;     // h1 dead after LN1

    // 0) one-time fp32 -> bf16 conversions (weights + x)
    auto conv = [&](const float* in, unsigned short* out, size_t n) {
        int n4 = (int)(n / 4);
        f32_to_bf16<<<dim3((n4 + 255) / 256), 256, 0, stream>>>((const float4*)in, (uint2*)out, n4);
    };
    conv(x,  x_bf,  (size_t)M * HID);
    conv(Wv, Wv_bf, (size_t)HID * HID);
    conv(Wm, Wm_bf, (size_t)HID * HID);
    conv(W1, W1_bf, (size_t)FF * HID);
    conv(W2, W2_bf, (size_t)FF * HID);

    // 1) p = x @ Wv^T + bv  -> bf16
    gemm_bf16_wmma<<<dim3(HID / 128, M / 128), 256, 0, stream>>>(
        x_bf, Wv_bf, bv, nullptr, nullptr, p_bf, M, HID, HID, 0);
    // 2) attention (q=k=v=p) -> bf16 (overwrites x_bf, which is dead)
    attn_kernel<<<dim3(S / 64, B * NH), 128, 0, stream>>>(p_bf, mask, att_bf);
    // 3) h1 = att @ Wm^T + bm + x
    gemm_bf16_wmma<<<dim3(HID / 128, M / 128), 256, 0, stream>>>(
        att_bf, Wm_bf, bm, x, h1, nullptr, M, HID, HID, 0);
    // 4) x1 = LN(h1)  (f32 + bf16)
    ln_kernel<<<dim3(M), 256, 0, stream>>>(h1, g1, be1, x1, x1_bf);
    // 5) ff1 = relu(x1 @ W1^T + b1) -> bf16
    gemm_bf16_wmma<<<dim3(FF / 128, M / 128), 256, 0, stream>>>(
        x1_bf, W1_bf, b1, nullptr, nullptr, ff1, M, FF, HID, 1);
    // 6) h2 = ff1 @ W2^T + b2 + x1   (reuses h1 buffer)
    gemm_bf16_wmma<<<dim3(HID / 128, M / 128), 256, 0, stream>>>(
        ff1, W2_bf, b2, x1, h2, nullptr, M, HID, FF, 0);
    // 7) out = LN(h2)
    ln_kernel<<<dim3(M), 256, 0, stream>>>(h2, g2, be2, (float*)d_out, nullptr);
}
